// ModelSimple_11596411699488
// MI455X (gfx1250) — matrix operand order repaired and checked by reading, not compiled, not run on hardware
//
#include <hip/hip_runtime.h>
#include <hip/hip_bf16.h>

typedef __bf16 bf16_t;
typedef __attribute__((ext_vector_type(16))) __bf16 v16bf;
typedef __attribute__((ext_vector_type(8)))  float  v8f;

// ---- problem sizes ----
#define B_       256
#define IN_SP    104976   // 18^4
#define OUT1_SP  20736    // 12^4
#define OUT2_SP  1296     // 6^4
#define K1       2401     // 7^4
#define K1P      2432     // padded to multiple of 32
#define NK1      76       // K1P/32
#define K2P      12160    // 5 * K1P
#define NK2      380      // K2P/32
#define DENSE_K  12960    // 10 * 6^4

// ---- workspace layout (bytes) ----
#define KOFF1_OFF 0u                        //  2432 * 4  =   9,728 B
#define KOFF2_OFF 16384u                    // 12160 * 4  =  48,640 B
#define W1S_OFF   81920u                    // 76*512*2   =  77,824 B
#define W2S_OFF   163840u                   // 380*512*2  = 389,120 B
#define H1_OFF    1048576u                  // 256*5*20736*2 = 53,084,160 B (bf16)
#define H2_OFF    (H1_OFF + 56u*1024u*1024u)// 256*12960*4 = 13,271,040 B (f32)

// im2col K-offset tables (spatial offsets within input volume)
__global__ void build_tables(int* __restrict__ koff1, int* __restrict__ koff2) {
    for (int k = threadIdx.x; k < K1P; k += blockDim.x) {
        int off = 0;
        if (k < K1) {
            int kx = k / 343, r = k % 343;
            int ky = r / 49;  r %= 49;
            int kz = r / 7,  kw = r % 7;
            off = kx * 5832 + ky * 324 + kz * 18 + kw;   // strides of 18^4 input
        }
        koff1[k] = off;
    }
    for (int k = threadIdx.x; k < K2P; k += blockDim.x) {
        int cc = k / 2432, kk = k % 2432;
        int off = 0;
        if (kk < K1) {
            int kx = kk / 343, r = kk % 343;
            int ky = r / 49;  r %= 49;
            int kz = r / 7,  kw = r % 7;
            off = cc * OUT1_SP + kx * 1728 + ky * 144 + kz * 12 + kw; // 12^4 input
        }
        koff2[k] = off;
    }
}

// K index inside a 32-wide chunk in the contiguous k order this file uses for both A and B:
// lanes 0-15: elems 0..15 -> K 0..15
// lanes16-31: elems 0..15 -> K 16..31
__device__ __forceinline__ int a_k_of(int lane, int i) {
    return ((lane >> 4) << 4) + i;
}

// swizzle conv weights into [kchunk][lane][16] bf16 so each lane's A fragment
// is one contiguous 32B load per K-step
__global__ void swizzle_w1(const float* __restrict__ w1, bf16_t* __restrict__ w1s) {
    int idx = blockIdx.x * blockDim.x + threadIdx.x;
    if (idx >= NK1 * 512) return;
    int kc = idx >> 9, lane = (idx >> 4) & 31, i = idx & 15;
    int M = lane & 15;
    int K = kc * 32 + a_k_of(lane, i);
    float v = 0.f;
    if (M < 5 && K < K1) v = w1[M * K1 + K];
    w1s[idx] = (bf16_t)v;
}

__global__ void swizzle_w2(const float* __restrict__ w2, bf16_t* __restrict__ w2s) {
    int idx = blockIdx.x * blockDim.x + threadIdx.x;
    if (idx >= NK2 * 512) return;
    int kc = idx >> 9, lane = (idx >> 4) & 31, i = idx & 15;
    int M = lane & 15;
    int K = kc * 32 + a_k_of(lane, i);
    int c = K / 2432, kk = K % 2432;
    float v = 0.f;
    if (M < 10 && kk < K1) v = w2[(M * 5 + c) * K1 + kk];
    w2s[idx] = (bf16_t)v;
}

// conv1: x[256,1,18^4] fp32 -> h1[256,5,12^4] bf16 (bias+relu fused)
// One wave computes TWO 16(Cout-pad) x 16(positions) tiles over K=2432:
// the A (weights) fragment is shared, the two WMMAs are independent and
// can pipeline back-to-back in the matrix unit.
__global__ __launch_bounds__(256) void conv1_wmma(
    const float* __restrict__ x, const bf16_t* __restrict__ w1s,
    const int* __restrict__ koff1, const float* __restrict__ b1,
    bf16_t* __restrict__ h1) {
    int wid  = (blockIdx.x * blockDim.x + threadIdx.x) >> 5;
    int lane = threadIdx.x & 31;
    int nWaves = (B_ * OUT1_SP) >> 5;   // 165,888 (exact)
    if (wid >= nWaves) return;          // wave-uniform

    int p0 = (wid << 5) + (lane & 15);  // first tile's position for this N lane
    int p1 = p0 + 16;                   // second tile

    int b0  = p0 / OUT1_SP, sp0 = p0 % OUT1_SP;
    int ox = sp0 / 1728, r = sp0 % 1728;
    int oy = r / 144;  r %= 144;
    int oz = r / 12, ow = r % 12;
    int base0 = b0 * IN_SP + ((ox * 18 + oy) * 18 + oz) * 18 + ow;

    int b1i = p1 / OUT1_SP, sp1 = p1 % OUT1_SP;
    ox = sp1 / 1728; r = sp1 % 1728;
    oy = r / 144;  r %= 144;
    oz = r / 12; ow = r % 12;
    int base1 = b1i * IN_SP + ((ox * 18 + oy) * 18 + oz) * 18 + ow;

    int khalf = (lane >> 4) << 4;       // B-fragment: lanes 0-15 K0..15, 16-31 K16..31

    v8f acc0 = {}, acc1 = {};
    for (int kc = 0; kc < NK1; ++kc) {
        v16bf a = *(const v16bf*)(w1s + kc * 512 + lane * 16);
        const int* kp = koff1 + kc * 32 + khalf;
        v16bf bf0, bf1;
#pragma unroll
        for (int i = 0; i < 16; ++i) {
            int o = kp[i];
            bf0[i] = (bf16_t)x[base0 + o];
            bf1[i] = (bf16_t)x[base1 + o];
        }
        acc0 = __builtin_amdgcn_wmma_f32_16x16x32_bf16(
            false, a, false, bf0, (short)0, acc0, false, false);
        acc1 = __builtin_amdgcn_wmma_f32_16x16x32_bf16(
            false, a, false, bf1, (short)0, acc1, false, false);
    }

    // C layout: lanes 0-15 hold M=0..7 in acc[0..7]; only M<5 valid for conv1
    if (lane < 16) {
#pragma unroll
        for (int m = 0; m < 5; ++m) {
            float bias = b1[m];
            float v0 = acc0[m] + bias;  v0 = v0 > 0.f ? v0 : 0.f;
            float v1 = acc1[m] + bias;  v1 = v1 > 0.f ? v1 : 0.f;
            h1[(b0  * 5 + m) * OUT1_SP + sp0] = (bf16_t)v0;
            h1[(b1i * 5 + m) * OUT1_SP + sp1] = (bf16_t)v1;
        }
    }
}

// conv2: h1[256,5,12^4] bf16 -> h2[256,10,6^4] fp32 (bias+relu fused)
__global__ __launch_bounds__(256) void conv2_wmma(
    const bf16_t* __restrict__ h1, const bf16_t* __restrict__ w2s,
    const int* __restrict__ koff2, const float* __restrict__ b2,
    float* __restrict__ h2) {
    int wid  = (blockIdx.x * blockDim.x + threadIdx.x) >> 5;
    int lane = threadIdx.x & 31;
    int nWaves = (B_ * OUT2_SP) >> 5;   // 10,368 (exact)
    if (wid >= nWaves) return;

    int p0 = (wid << 5) + (lane & 15);
    int p1 = p0 + 16;

    int b0  = p0 / OUT2_SP, sp0 = p0 % OUT2_SP;
    int ox = sp0 / 216, r = sp0 % 216;
    int oy = r / 36;  r %= 36;
    int oz = r / 6, ow = r % 6;
    int base0 = b0 * 5 * OUT1_SP + ((ox * 12 + oy) * 12 + oz) * 12 + ow;

    int b1i = p1 / OUT2_SP, sp1 = p1 % OUT2_SP;
    ox = sp1 / 216; r = sp1 % 216;
    oy = r / 36;  r %= 36;
    oz = r / 6; ow = r % 6;
    int base1 = b1i * 5 * OUT1_SP + ((ox * 12 + oy) * 12 + oz) * 12 + ow;

    int khalf = (lane >> 4) << 4;

    v8f acc0 = {}, acc1 = {};
    for (int kc = 0; kc < NK2; ++kc) {
        v16bf a = *(const v16bf*)(w2s + kc * 512 + lane * 16);
        const int* kp = koff2 + kc * 32 + khalf;
        v16bf bf0, bf1;
#pragma unroll
        for (int i = 0; i < 16; ++i) {
            int o = kp[i];
            bf0[i] = h1[base0 + o];
            bf1[i] = h1[base1 + o];
        }
        acc0 = __builtin_amdgcn_wmma_f32_16x16x32_bf16(
            false, a, false, bf0, (short)0, acc0, false, false);
        acc1 = __builtin_amdgcn_wmma_f32_16x16x32_bf16(
            false, a, false, bf1, (short)0, acc1, false, false);
    }

    // lanes 0-15: M=0..7; lanes 16-31: M=8..15 (only 8,9 valid)
    int mbase = (lane >> 4) ? 8 : 0;
#pragma unroll
    for (int rr = 0; rr < 8; ++rr) {
        int m = mbase + rr;
        if (m < 10) {
            float bias = b2[m];
            float v0 = acc0[rr] + bias;  v0 = v0 > 0.f ? v0 : 0.f;
            float v1 = acc1[rr] + bias;  v1 = v1 > 0.f ? v1 : 0.f;
            h2[b0  * DENSE_K + m * OUT2_SP + sp0] = v0;  // matches reshape(B,-1)
            h2[b1i * DENSE_K + m * OUT2_SP + sp1] = v1;
        }
    }
}

// dense [256,12960] @ [2,12960]^T + bias, then 2-way softmax
__global__ __launch_bounds__(256) void dense_softmax(
    const float* __restrict__ h2, const float* __restrict__ wd,
    const float* __restrict__ bd, float* __restrict__ out) {
    __shared__ float s0[256], s1[256];
    int b = blockIdx.x, tid = threadIdx.x;
    const float* h = h2 + b * DENSE_K;
    float a0 = 0.f, a1 = 0.f;
    for (int k = tid; k < DENSE_K; k += 256) {
        float v = h[k];
        a0 += v * wd[k];
        a1 += v * wd[DENSE_K + k];
    }
    s0[tid] = a0; s1[tid] = a1;
    __syncthreads();
    for (int s = 128; s > 0; s >>= 1) {
        if (tid < s) { s0[tid] += s0[tid + s]; s1[tid] += s1[tid + s]; }
        __syncthreads();
    }
    if (tid == 0) {
        float l0 = s0[0] + bd[0], l1 = s1[0] + bd[1];
        float m  = fmaxf(l0, l1);
        float e0 = __expf(l0 - m), e1 = __expf(l1 - m);
        float inv = 1.f / (e0 + e1);
        out[b * 2 + 0] = e0 * inv;
        out[b * 2 + 1] = e1 * inv;
    }
}

extern "C" void kernel_launch(void* const* d_in, const int* in_sizes, int n_in,
                              void* d_out, int out_size, void* d_ws, size_t ws_size,
                              hipStream_t stream) {
    (void)in_sizes; (void)n_in; (void)out_size; (void)ws_size;
    const float* x  = (const float*)d_in[0];
    const float* w1 = (const float*)d_in[1];
    const float* b1 = (const float*)d_in[2];
    const float* w2 = (const float*)d_in[3];
    const float* b2 = (const float*)d_in[4];
    const float* wd = (const float*)d_in[5];
    const float* bd = (const float*)d_in[6];
    float* out = (float*)d_out;

    char* ws = (char*)d_ws;
    int*    koff1 = (int*)(ws + KOFF1_OFF);
    int*    koff2 = (int*)(ws + KOFF2_OFF);
    bf16_t* w1s   = (bf16_t*)(ws + W1S_OFF);
    bf16_t* w2s   = (bf16_t*)(ws + W2S_OFF);
    bf16_t* h1    = (bf16_t*)(ws + H1_OFF);
    float*  h2    = (float*)(ws + H2_OFF);

    build_tables<<<1, 256, 0, stream>>>(koff1, koff2);
    swizzle_w1<<<(NK1 * 512 + 255) / 256, 256, 0, stream>>>(w1, w1s);
    swizzle_w2<<<(NK2 * 512 + 255) / 256, 256, 0, stream>>>(w2, w2s);

    // conv1: 165,888 dual-tile waves, 8 waves (256 threads) per block
    conv1_wmma<<<(B_ * OUT1_SP / 32) / 8, 256, 0, stream>>>(x, w1s, koff1, b1, h1);
    // conv2: 10,368 dual-tile waves
    conv2_wmma<<<(B_ * OUT2_SP / 32) / 8, 256, 0, stream>>>(h1, w2s, koff2, b2, h2);

    dense_softmax<<<B_, 256, 0, stream>>>(h2, wd, bd, out);
}
